// KANLayer_28862180229294
// MI455X (gfx1250) — matrix operand order repaired and checked by reading, not compiled, not run on hardware
//
#include <hip/hip_runtime.h>

// ---------------------------------------------------------------------------
// KAN layer on gfx1250:
//   out[b,o] = sum_i mask*( scale_base*silu(x) + scale_sp*sum_k B[b,i,k]*coef )
// restructured as a single bf16 WMMA GEMM:  out = F(4096x2304) @ W(2304x256)
//   features per input dim: 8 cubic B-spline basis values + 1 SiLU value
// ---------------------------------------------------------------------------

#define BATCH   4096
#define IN_DIM  256
#define OUT_DIM 256
#define NK      8      // NUM + K  (spline coefficients per (i,o))
#define FK      9      // features per input dim (8 spline + 1 base)
#define GPTS    12     // grid points per input dim (NUM+1+2K)
#define KEFF    (IN_DIM * FK)   // 2304 = 72 * 32
#define CHUNK_I 32              // input dims per LDS chunk
#define CHUNK_K (CHUNK_I * FK)  // 288 = 9 * 32
#define ROWS    32              // batch rows per workgroup
#define NTILE   4               // 16x16 N tiles per wave (64 cols)

typedef __attribute__((ext_vector_type(16))) __bf16 v16bf;
typedef __attribute__((ext_vector_type(8)))  __bf16 v8bf;
typedef __attribute__((ext_vector_type(8)))  float  v8f;

// ---------------------------------------------------------------------------
// Kernel 1: fold coef/scales/mask into bf16 Wt[o][k], o-major, k = i*9 + kk.
// One block per output column o; one thread per input dim i.
// ---------------------------------------------------------------------------
__global__ __launch_bounds__(256)
void kan_prep_w(const float* __restrict__ coef,
                const float* __restrict__ scale_base,
                const float* __restrict__ scale_sp,
                const float* __restrict__ mask,
                unsigned short* __restrict__ WtRaw)
{
    __bf16* Wt = reinterpret_cast<__bf16*>(WtRaw);
    const int o = blockIdx.x;     // 0..255
    const int i = threadIdx.x;    // 0..255
    const int io = i * OUT_DIM + o;
    const float m  = mask[io];
    const float sb = scale_base[io] * m;
    const float ss = scale_sp[io]   * m;
    __bf16* dst = Wt + (size_t)o * KEFF + (size_t)i * FK;
    const float* cp = coef + (size_t)io * NK;
#pragma unroll
    for (int k = 0; k < NK; ++k)
        dst[k] = (__bf16)(ss * cp[k]);
    dst[NK] = (__bf16)sb;
}

// ---------------------------------------------------------------------------
// Kernel 2: fused feature build + bf16 WMMA GEMM.
// Block = 256 threads (8 wave32s). Each block: 32 batch rows x 256 out cols
// -> 128 blocks (fills the WGP array; 18 KB LDS allows 2 WGs/WGP).
//   wave_m = wave&1 -> 16-row M tile, wave_n = wave>>1 -> 64-col quarter.
//   Each wave: 4 accumulators (32 VGPRs) + 4 B-fragments (32 VGPRs).
// K loop: 8 chunks of 32 input dims (288 K values) staged in LDS as bf16,
// then 9 WMMA K-steps of 32 per chunk.
//
// Fill phase: thread t owns input dim i = i0 + (t&31) for the whole chunk,
// so the grid row and the 30 guarded knot-spacing reciprocals are computed
// ONCE per chunk, and the 4 batch rows use multiply-only Cox-de Boor.
//
// GEMM phase: the 4 B-fragment loads are fenced from the 4 WMMAs with
// sched_barrier(0) so all 8 global_load_b128 stay in flight together
// (the round-3 codegen had collapsed them onto one register pair with
// s_wait_loadcnt 0x0 per tile -> MLP of 1).
// ---------------------------------------------------------------------------
__global__ __launch_bounds__(256)
void kan_main(const float* __restrict__ x,
              const float* __restrict__ grid,
              const unsigned short* __restrict__ WtRaw,
              float* __restrict__ out)
{
    const __bf16* Wt = reinterpret_cast<const __bf16*>(WtRaw);
    __shared__ __bf16 ldsF[ROWS][CHUNK_K];   // 32 rows x 288 bf16 = 18 KB

    const int tid    = threadIdx.x;
    const int lane   = tid & 31;
    const int wave   = tid >> 5;
    const int wave_m = wave & 1;           // 0..1  -> 16-row subtile
    const int wave_n = wave >> 1;          // 0..3  -> 64-col quarter
    const int r0     = blockIdx.x * ROWS;  // batch row base

    const v8f vzero = {0.f, 0.f, 0.f, 0.f, 0.f, 0.f, 0.f, 0.f};
    v8f acc[NTILE];
#pragma unroll
    for (int t = 0; t < NTILE; ++t) acc[t] = vzero;

    // Fragment addressing (contiguous k order chosen here for both A and B, not the ISA table order; wave32):
    //   A 16x32: lane<16 holds K{0..15},        lane>=16 holds K{16..31}
    //   B 32x16: lane<16 holds K{0..15},        lane>=16 holds K{16..31}
    const int mrow   = 16 * wave_m + (lane & 15);
    const int kbaseA = (lane >> 4) * 16;
    const int koffB  = (lane >> 4) * 16;

    // Per-wave B-fragment base pointer: column n = 64*wave_n + 16*nt + (lane&15).
    const __bf16* WtLane = Wt + (size_t)(64 * wave_n + (lane & 15)) * KEFF + koffB;

    // Fill-phase ownership: fixed input-dim slot, 4 batch rows per chunk.
    const int ii      = tid & 31;          // input-dim slot within chunk
    const int rowbase = tid >> 5;          // 0..7 ; rows rowbase + 8*q, q=0..3

    for (int c = 0; c < 8; ++c) {
        const int i = CHUNK_I * c + ii;    // global input dim for this thread

        // ---- per-chunk invariants: grid row + guarded reciprocals ----
        float gv[GPTS];
        const float* g = grid + (size_t)i * GPTS;
#pragma unroll
        for (int j = 0; j < GPTS; ++j) gv[j] = g[j];

        // r_p[j] = 1/(g[j+p]-g[j]) (0 if degenerate), j = 0..11-p
        float r1[11], r2[10], r3[9];
#pragma unroll
        for (int j = 0; j < 11; ++j) {
            float d = gv[j + 1] - gv[j];
            r1[j] = (d != 0.0f) ? 1.0f / d : 0.0f;
        }
#pragma unroll
        for (int j = 0; j < 10; ++j) {
            float d = gv[j + 2] - gv[j];
            r2[j] = (d != 0.0f) ? 1.0f / d : 0.0f;
        }
#pragma unroll
        for (int j = 0; j < 9; ++j) {
            float d = gv[j + 3] - gv[j];
            r3[j] = (d != 0.0f) ? 1.0f / d : 0.0f;
        }

        // ---- 4 batch rows: multiply-only Cox-de Boor + SiLU -> LDS ----
#pragma unroll
        for (int q = 0; q < 4; ++q) {
            const int row = rowbase + 8 * q;           // 0..31
            const float xv = x[(size_t)(r0 + row) * IN_DIM + i];

            float B[11];
#pragma unroll
            for (int j = 0; j < 11; ++j)
                B[j] = (xv >= gv[j] && xv < gv[j + 1]) ? 1.0f : 0.0f;
#pragma unroll
            for (int j = 0; j < 10; ++j)
                B[j] = (xv - gv[j]) * r1[j] * B[j]
                     + (gv[j + 2] - xv) * r1[j + 1] * B[j + 1];
#pragma unroll
            for (int j = 0; j < 9; ++j)
                B[j] = (xv - gv[j]) * r2[j] * B[j]
                     + (gv[j + 3] - xv) * r2[j + 1] * B[j + 1];
#pragma unroll
            for (int j = 0; j < 8; ++j)
                B[j] = (xv - gv[j]) * r3[j] * B[j]
                     + (gv[j + 4] - xv) * r3[j + 1] * B[j + 1];

            __bf16* dst = &ldsF[row][ii * FK];
#pragma unroll
            for (int k = 0; k < NK; ++k) dst[k] = (__bf16)B[k];
            dst[NK] = (__bf16)(xv * (1.0f / (1.0f + __expf(-xv))));   // SiLU
        }
        __syncthreads();

        // ---- 9 WMMA K-steps over this 288-wide chunk ----
        for (int kt = 0; kt < 9; ++kt) {
            const int kb = kt * 32;
            const int gk = c * CHUNK_K + kb;

            // A fragment from LDS (two ds_load_b128 + lane-half shuffle).
            v8bf alo = *(const v8bf*)&ldsF[mrow][kb + kbaseA];
            v8bf ahi = *(const v8bf*)&ldsF[mrow][kb + 8 + kbaseA];
            v16bf afrag = __builtin_shufflevector(alo, ahi,
                0, 1, 2, 3, 4, 5, 6, 7, 8, 9, 10, 11, 12, 13, 14, 15);

            // Load ALL 4 B fragments -> 8 global_load_b128 in flight.
            v16bf bf[NTILE];
#pragma unroll
            for (int nt = 0; nt < NTILE; ++nt)
                bf[nt] = *(const v16bf*)(WtLane + (size_t)(16 * nt) * KEFF + gk);

            // Fence: keep all fragments live so loads overlap, then drain.
            __builtin_amdgcn_sched_barrier(0);

#pragma unroll
            for (int nt = 0; nt < NTILE; ++nt)
                acc[nt] = __builtin_amdgcn_wmma_f32_16x16x32_bf16(
                    false, afrag, false, bf[nt], (short)0, acc[nt], false, false);
        }
        __syncthreads();
    }

    // ---- epilogue: C/D layout -> lane<16: M=r, lane>=16: M=8+r; N=lane&15 ----
    const int rbase = r0 + 16 * wave_m + (lane >> 4) * 8;
#pragma unroll
    for (int nt = 0; nt < NTILE; ++nt) {
        const int col = 64 * wave_n + 16 * nt + (lane & 15);
#pragma unroll
        for (int r = 0; r < 8; ++r)
            out[(size_t)(rbase + r) * OUT_DIM + col] = acc[nt][r];
    }
}

// ---------------------------------------------------------------------------
extern "C" void kernel_launch(void* const* d_in, const int* in_sizes, int n_in,
                              void* d_out, int out_size, void* d_ws, size_t ws_size,
                              hipStream_t stream) {
    const float* x          = (const float*)d_in[0];  // (4096, 256)
    const float* grid       = (const float*)d_in[1];  // (256, 12)
    const float* coef       = (const float*)d_in[2];  // (256, 256, 8)
    const float* scale_base = (const float*)d_in[3];  // (256, 256)
    const float* scale_sp   = (const float*)d_in[4];  // (256, 256)
    const float* mask       = (const float*)d_in[5];  // (256, 256)
    float* out = (float*)d_out;                       // (4096, 256)

    unsigned short* Wt = (unsigned short*)d_ws;       // 2304*256 bf16 = 1.13 MB

    (void)in_sizes; (void)n_in; (void)out_size; (void)ws_size;

    kan_prep_w<<<dim3(OUT_DIM), dim3(256), 0, stream>>>(coef, scale_base,
                                                        scale_sp, mask, Wt);
    kan_main<<<dim3(BATCH / ROWS), dim3(256), 0, stream>>>(x, grid, Wt, out);
}
